// SeqAttention_23441931501655
// MI455X (gfx1250) — matrix operand order repaired and checked by reading, not compiled, not run on hardware
//
#include <hip/hip_runtime.h>
#include <hip/hip_bf16.h>

typedef __bf16 bf16_t;
typedef __bf16 v16bf __attribute__((ext_vector_type(16)));
typedef float  v8f   __attribute__((ext_vector_type(8)));

#define Mdim   1024
#define Ldim   1024
#define Ddim   64
#define MLdim  2048
#define WAVES  4
#define ROWS_WG 64          // 4 waves x 16 rows
#define NKT    34           // 32-key tiles covering 64+L-1 = 1087 keys

// LDS partition (bf16 elements)
#define P_ELEMS    (WAVES * 16 * Ldim)      // 65536 : per-wave P strips
#define STAGE_OFF  (P_ELEMS)                // 4096  : PET chunk (phase1) / K+VT (phase2)
#define STAGE_ELEMS 4096
#define PR_OFF     (STAGE_OFF + STAGE_ELEMS)
#define PR_ELEMS   (WAVES * 16 * 32)        // 2048
#define SMEM_BYTES ((PR_OFF + PR_ELEMS) * 2)  // 143360 B

__device__ __forceinline__ unsigned pack2(float a, float b) {
  union { bf16_t h[2]; unsigned u; } x;
  x.h[0] = (bf16_t)a; x.h[1] = (bf16_t)b;
  return x.u;
}

__device__ __forceinline__ v8f wmma_bf16(v16bf a, v16bf b, v8f c) {
  return __builtin_amdgcn_wmma_f32_16x16x32_bf16(false, a, false, b,
                                                 (short)0, c, false, false);
}

// 4 waves per block; wave w owns query rows [i0wg + 16w, i0wg + 16w + 16).
__global__ __launch_bounds__(128)
void seqattn_kernel(const float* __restrict__ q,
                    const float* __restrict__ k,
                    const float* __restrict__ v,
                    const float* __restrict__ pe,
                    float* __restrict__ out) {
  extern __shared__ bf16_t smem[];
  const int tid  = threadIdx.x;
  const int wave = tid >> 5;
  const int lane = tid & 31;
  const int lh   = lane >> 4;
  const int ln   = lane & 15;
  const int b    = blockIdx.y;
  const int i0wg = blockIdx.x * ROWS_WG;
  const int i0   = i0wg + wave * 16;

  bf16_t* Plds  = smem + wave * 16 * Ldim;   // this wave's P strip [16][1024]
  bf16_t* stage = smem + STAGE_OFF;          // shared staging (8KB)
  bf16_t* Klds  = stage;                     // [32][64] bf16 (phase 2)
  bf16_t* VTlds = stage + 32 * Ddim;         // [64][32] bf16, transposed V
  bf16_t* probs = smem + PR_OFF + wave * 16 * 32;

  const float* qb = q + ((size_t)b * Mdim + i0) * Ddim;
  const float* kb = k + (size_t)b * MLdim * Ddim;
  const float* vb = v + (size_t)b * MLdim * Ddim;

  // ---- Q A-fragments (16x32 bf16): A(m,kk) -> lane = m + 16*(kk/16),
  //      element s = kk%16  => per lane kk = 16*lh + s
  v16bf qa[2];
  {
    const float* qrow = qb + (size_t)ln * Ddim;
#pragma unroll
    for (int kb2 = 0; kb2 < 2; ++kb2) {
#pragma unroll
      for (int s = 0; s < 16; s += 2) {
        int vv = s >> 1;
        int d  = 32 * kb2 + 16 * lh + 2 * vv;
        float2 f = *(const float2*)(qrow + d);
        qa[kb2][s]     = (bf16_t)f.x;
        qa[kb2][s + 1] = (bf16_t)f.y;
      }
    }
  }

  // ---- Phase 1: P[m][j] = q[i0+m].pe[:,j] in 64-col chunks via staged PE^T ----
  for (int c = 0; c < Ldim / 64; ++c) {
    const int j0 = c * 64;
    __syncthreads();
    {   // stage PET[jl][d] (bf16) for j = j0..j0+63 ; coalesced along j
      const int d   = tid >> 1;
      const int jl0 = (tid & 1) * 32;
      const float* src = pe + (size_t)d * Ldim + j0 + jl0;
#pragma unroll
      for (int s = 0; s < 32; s += 4) {
        float4 f = *(const float4*)(src + s);
        stage[(jl0 + s + 0) * Ddim + d] = (bf16_t)f.x;
        stage[(jl0 + s + 1) * Ddim + d] = (bf16_t)f.y;
        stage[(jl0 + s + 2) * Ddim + d] = (bf16_t)f.z;
        stage[(jl0 + s + 3) * Ddim + d] = (bf16_t)f.w;
      }
    }
    __syncthreads();
#pragma unroll
    for (int jt = 0; jt < 4; ++jt) {
      v8f acc = {};
#pragma unroll
      for (int kb2 = 0; kb2 < 2; ++kb2) {
        // B(kk,n): lane n=ln, kk = 16*lh + s  => d contiguous from 32*kb2+16*lh
        v16bf pb = *(const v16bf*)(stage + (jt * 16 + ln) * Ddim + 32 * kb2 + 16 * lh);
        acc = wmma_bf16(qa[kb2], pb, acc);
      }
#pragma unroll
      for (int r = 0; r < 8; ++r)               // C layout: m = r + 8*lh, n = ln
        Plds[(r + 8 * lh) * Ldim + j0 + jt * 16 + ln] = (bf16_t)acc[r];
    }
  }

  // ---- Phase 2: flash attention over 34 key tiles of 32 keys ----
  float mrow[8], lrow[8];
  v8f oacc[4];
#pragma unroll
  for (int r = 0; r < 8; ++r) { mrow[r] = -1e30f; lrow[r] = 0.0f; }
#pragma unroll
  for (int nt = 0; nt < 4; ++nt)
#pragma unroll
    for (int r = 0; r < 8; ++r) oacc[nt][r] = 0.0f;

  for (int t = 0; t < NKT; ++t) {
    const int kt0 = i0wg + 32 * t;            // in-bounds: kt0+31 <= 2047 always
    __syncthreads();                          // previous tile's LDS reads done
    {   // stage K [32][64] and V^T [64][32] as bf16; 16 f32 per thread each
      const int row = tid >> 2;
      const int dc  = (tid & 3) << 4;
      const float* gk = kb + (size_t)(kt0 + row) * Ddim + dc;
      const float* gv = vb + (size_t)(kt0 + row) * Ddim + dc;
      unsigned* kst = (unsigned*)(Klds + row * Ddim + dc);
#pragma unroll
      for (int s4 = 0; s4 < 4; ++s4) {
        float4 f = *(const float4*)(gk + s4 * 4);
        kst[s4 * 2 + 0] = pack2(f.x, f.y);
        kst[s4 * 2 + 1] = pack2(f.z, f.w);
      }
#pragma unroll
      for (int s4 = 0; s4 < 4; ++s4) {
        float4 g = *(const float4*)(gv + s4 * 4);
        VTlds[(dc + s4 * 4 + 0) * 32 + row] = (bf16_t)g.x;
        VTlds[(dc + s4 * 4 + 1) * 32 + row] = (bf16_t)g.y;
        VTlds[(dc + s4 * 4 + 2) * 32 + row] = (bf16_t)g.z;
        VTlds[(dc + s4 * 4 + 3) * 32 + row] = (bf16_t)g.w;
      }
    }
    __syncthreads();

    // content scores: two 16x16 n-tiles, B-fragments straight from Klds
    v8f sc[2];
#pragma unroll
    for (int nt = 0; nt < 2; ++nt) {
      v8f acc = {};
#pragma unroll
      for (int kb2 = 0; kb2 < 2; ++kb2) {
        v16bf kf = *(const v16bf*)(Klds + (nt * 16 + ln) * Ddim + 32 * kb2 + 16 * lh);
        acc = wmma_bf16(qa[kb2], kf, acc);
      }
      sc[nt] = acc;
    }

    // logits = (where(cont==0,-1e6,cont) + P[m][j]) / 8 inside band 0<=j<L
    float lgs[2][8];
    float tmax[8];
#pragma unroll
    for (int r = 0; r < 8; ++r) tmax[r] = -1e30f;
#pragma unroll
    for (int nt = 0; nt < 2; ++nt) {
#pragma unroll
      for (int r = 0; r < 8; ++r) {
        int m = r + 8 * lh;
        int j = 32 * t + nt * 16 + ln - wave * 16 - m;   // k - i
        float cont = sc[nt][r];
        if (cont == 0.0f) cont = -1e6f;
        float lg = -1e30f;
        if (j >= 0 && j < Ldim) {
          float pos = (float)Plds[m * Ldim + j];
          lg = (cont + pos) * 0.125f;                    // 1/sqrt(64)
        }
        lgs[nt][r] = lg;
        tmax[r] = fmaxf(tmax[r], lg);
      }
    }
#pragma unroll
    for (int r = 0; r < 8; ++r) {       // row max across 16-lane halves
      float x = tmax[r];
#pragma unroll
      for (int off = 1; off < 16; off <<= 1)
        x = fmaxf(x, __shfl_xor(x, off, 32));
      tmax[r] = x;
    }

    float alpha[8], rsum[8];
#pragma unroll
    for (int r = 0; r < 8; ++r) {
      float mnew = fmaxf(mrow[r], tmax[r]);
      alpha[r] = __expf(mrow[r] - mnew);
      mrow[r]  = mnew;
      rsum[r]  = 0.0f;
    }
#pragma unroll
    for (int nt = 0; nt < 2; ++nt) {
#pragma unroll
      for (int r = 0; r < 8; ++r) {
        int m = r + 8 * lh;
        float lg = lgs[nt][r];
        float p  = (lg > -1e29f) ? __expf(lg - mrow[r]) : 0.0f;
        rsum[r] += p;
        probs[m * 32 + nt * 16 + ln] = (bf16_t)p;   // wave-private buffer
      }
    }
#pragma unroll
    for (int r = 0; r < 8; ++r) {
      float x = rsum[r];
#pragma unroll
      for (int off = 1; off < 16; off <<= 1)
        x += __shfl_xor(x, off, 32);
      lrow[r] = lrow[r] * alpha[r] + x;
    }

    // in-wave C->A layout roundtrip: DS ops are in-order per wave; fence the compiler
    asm volatile("s_wait_dscnt 0" ::: "memory");
    v16bf pra;
#pragma unroll
    for (int s = 0; s < 16; ++s) {
      int vv = s >> 1;
      int kk = 16 * lh + 2 * vv + (s & 1);
      pra[s] = probs[ln * 32 + kk];
    }

    // O = O*alpha + Pr @ V ; V^T fragments are contiguous 32B LDS reads
#pragma unroll
    for (int nt = 0; nt < 4; ++nt) {
#pragma unroll
      for (int r = 0; r < 8; ++r) oacc[nt][r] *= alpha[r];
      v16bf vf = *(const v16bf*)(VTlds + (nt * 16 + ln) * 32 + 16 * lh);
      oacc[nt] = wmma_bf16(pra, vf, oacc[nt]);
    }
  }

  // ---- epilogue ----
  float* ob = out + ((size_t)b * Mdim + i0) * Ddim;
#pragma unroll
  for (int nt = 0; nt < 4; ++nt) {
#pragma unroll
    for (int r = 0; r < 8; ++r) {
      int m = r + 8 * lh;
      ob[(size_t)m * Ddim + nt * 16 + ln] = oacc[nt][r] / lrow[r];
    }
  }
}

extern "C" void kernel_launch(void* const* d_in, const int* in_sizes, int n_in,
                              void* d_out, int out_size, void* d_ws, size_t ws_size,
                              hipStream_t stream) {
  const float* q  = (const float*)d_in[0];   // (B, M, D)
  const float* k  = (const float*)d_in[1];   // (B, M+L, D)
  const float* v  = (const float*)d_in[2];   // (B, M+L, D)
  const float* pe = (const float*)d_in[3];   // (1, D, L)
  float* out = (float*)d_out;                // (B, M, D)

  const int B = in_sizes[0] / (Mdim * Ddim); // 64
  dim3 grid(Mdim / ROWS_WG, B);
  dim3 block(WAVES * 32);
  seqattn_kernel<<<grid, block, SMEM_BYTES, stream>>>(q, k, v, pe, out);
}
